// Model_49933289783864
// MI455X (gfx1250) — matrix operand order repaired and checked by reading, not compiled, not run on hardware
//
#include <hip/hip_runtime.h>
#include <hip/hip_bf16.h>

// ---------------------------------------------------------------------------
// Problem constants (match reference)
// ---------------------------------------------------------------------------
#define BATCH_SZ 65536
#define NF       512      // NUM_FEATURE (also fc0/fc1 output width)
#define HID      256      // HIDDEN
#define NC       10       // NUM_CLASSES
#define L1_COEF  1e-4
#define ALPHA    1e-3

typedef __attribute__((ext_vector_type(16))) __bf16 bf16x16;
typedef __attribute__((ext_vector_type(4)))  __bf16 bf16x4;
typedef __attribute__((ext_vector_type(8)))  float  v8f;

// ---------------------------------------------------------------------------
// 0) zero accumulators
// ---------------------------------------------------------------------------
__global__ void zero_kernel(float* __restrict__ sums, float* __restrict__ counts,
                            double* __restrict__ scal) {
    int t = blockIdx.x * 256 + threadIdx.x;
    for (int i = t; i < NC * NF; i += gridDim.x * 256) sums[i] = 0.f;
    if (blockIdx.x == 0) {
        if (threadIdx.x < NC) counts[threadIdx.x] = 0.f;
        if (threadIdx.x < 8)  scal[threadIdx.x]  = 0.0;
    }
}

// ---------------------------------------------------------------------------
// 1) transpose + convert f32 weight [K][N] -> bf16 [N][K], fused L1 |.| sum
// ---------------------------------------------------------------------------
__global__ __launch_bounds__(256)
void convT_l1_kernel(const float* __restrict__ src, __bf16* __restrict__ dst,
                     int K, int N, double* __restrict__ l1out) {
    __shared__ float red[256];
    float s = 0.f;
    const int total = K * N;
    for (int idx = blockIdx.x * 256 + threadIdx.x; idx < total; idx += gridDim.x * 256) {
        float v = src[idx];
        s += fabsf(v);
        int k = idx / N, n = idx % N;
        dst[(size_t)n * K + k] = (__bf16)v;
    }
    red[threadIdx.x] = s;
    __syncthreads();
    for (int o = 128; o > 0; o >>= 1) {
        if (threadIdx.x < o) red[threadIdx.x] += red[threadIdx.x + o];
        __syncthreads();
    }
    if (threadIdx.x == 0) atomicAdd(l1out, (double)red[0]);
}

// ---------------------------------------------------------------------------
// 2) L1 |.| reduction (for w1 only)
// ---------------------------------------------------------------------------
__global__ void l1_kernel(const float* __restrict__ p, int n, double* __restrict__ out) {
    __shared__ float red[256];
    float s = 0.f;
    for (int i = blockIdx.x * 256 + threadIdx.x; i < n; i += gridDim.x * 256)
        s += fabsf(p[i]);
    red[threadIdx.x] = s;
    __syncthreads();
    for (int o = 128; o > 0; o >>= 1) {
        if (threadIdx.x < o) red[threadIdx.x] += red[threadIdx.x + o];
        __syncthreads();
    }
    if (threadIdx.x == 0) atomicAdd(out, (double)red[0]);
}

// ---------------------------------------------------------------------------
// Cooperative A-tile staging: 16x32 f32 tile -> LDS in this code's contiguous-k A-fragment
// order (lane l: M = l&15, K-group g = l>>4; slot j -> K=16g+j).
// One float4 load + one bf16x4 LDS store per
// thread (threads 0..127), branch-free index math, no inner loop.
// ---------------------------------------------------------------------------
__device__ __forceinline__ void load_a_tile(__bf16 (*ldsA)[16],
                                            const float* __restrict__ base,
                                            int row0, int k0, int tid) {
    if (tid < 128) {
        const int m = tid >> 3;             // row in tile: 0..15
        const int c = (tid & 7) << 2;       // k offset: 0,4,...,28
        float4 f = *reinterpret_cast<const float4*>(
            base + (size_t)(row0 + m) * NF + k0 + c);
        const int g  = c >> 4;              // K-group 0/1
        const int j0 = c & 15;              // slot within group
        bf16x4 v;
        v[0] = (__bf16)f.x; v[1] = (__bf16)f.y;
        v[2] = (__bf16)f.z; v[3] = (__bf16)f.w;
        *reinterpret_cast<bf16x4*>(&ldsA[m + 16 * g][j0]) = v;
    }
}

// ---------------------------------------------------------------------------
// 3) fused gate block:  x4 = x + (1+sigmoid(x@fc0)) * relu(x@fc1)
//    grid: (BATCH/16, 4), block 256 (8 waves); wave -> 16 columns; both
//    GEMMs share the A fragment; B fragments software-pipelined.
// ---------------------------------------------------------------------------
__global__ __launch_bounds__(256)
void gemm_gate_kernel(const float* __restrict__ x,
                      const __bf16* __restrict__ fc0T,
                      const __bf16* __restrict__ fc1T,
                      float* __restrict__ x4) {
    __shared__ alignas(32) __bf16 ldsA[32][16];

    const int bt   = blockIdx.x;           // 16-row batch tile
    const int cg   = blockIdx.y;           // 128-col group
    const int tid  = threadIdx.x;
    const int lane = tid & 31;
    const int wv   = tid >> 5;
    const int n0   = cg * 128 + wv * 16;
    const int row0 = bt * 16;

    v8f acc0 = {};
    v8f acc1 = {};
    const __bf16* pB0 = fc0T + (size_t)(n0 + (lane & 15)) * NF + (lane >> 4) * 16;
    const __bf16* pB1 = fc1T + (size_t)(n0 + (lane & 15)) * NF + (lane >> 4) * 16;

    // prefetch first B fragments
    bf16x16 b0 = *reinterpret_cast<const bf16x16*>(pB0);
    bf16x16 b1 = *reinterpret_cast<const bf16x16*>(pB1);

    for (int k0 = 0; k0 < NF; k0 += 32) {
        __syncthreads();                        // protect previous A reads
        load_a_tile(ldsA, x, row0, k0, tid);
        __syncthreads();
        bf16x16 a = *reinterpret_cast<const bf16x16*>(&ldsA[lane][0]);
        bf16x16 nb0 = b0, nb1 = b1;
        if (k0 + 32 < NF) {                     // prefetch next B fragments
            nb0 = *reinterpret_cast<const bf16x16*>(pB0 + k0 + 32);
            nb1 = *reinterpret_cast<const bf16x16*>(pB1 + k0 + 32);
        }
        acc0 = __builtin_amdgcn_wmma_f32_16x16x32_bf16(false, a, false, b0, (short)0, acc0, false, false);
        acc1 = __builtin_amdgcn_wmma_f32_16x16x32_bf16(false, a, false, b1, (short)0, acc1, false, false);
        b0 = nb0; b1 = nb1;
    }

    const int half = lane >> 4;
    const int col  = n0 + (lane & 15);
#pragma unroll
    for (int r = 0; r < 8; ++r) {
        int m = r + 8 * half;
        size_t idx = (size_t)(row0 + m) * NF + col;
        float xv = x[idx];
        float x1 = acc0[r];
        float x2 = fmaxf(acc1[r], 0.f);
        float sg = 1.0f / (1.0f + expf(-x1));
        x4[idx] = xv + (1.0f + sg) * x2;
    }
}

// ---------------------------------------------------------------------------
// 4) head: h = relu(x4 @ w) [WMMA, 16x256 tile in LDS], logits = h @ w1,
//    then log-softmax / CE / argmax per row with f64 atomics.
//    grid: BATCH/16, block 256; wave -> 2 N-tiles (cols w*32 .. w*32+31).
// ---------------------------------------------------------------------------
__global__ __launch_bounds__(256)
void head_kernel(const float* __restrict__ x4,
                 const __bf16* __restrict__ wT,
                 const float* __restrict__ w1,
                 const int* __restrict__ y,
                 double* __restrict__ ce_sum,
                 double* __restrict__ acc_sum) {
    __shared__ alignas(32) __bf16 ldsA[32][16];
    __shared__ float sH[16][HID];     // 16 KB
    __shared__ float sL[16][NC];

    const int bt   = blockIdx.x;
    const int tid  = threadIdx.x;
    const int lane = tid & 31;
    const int wv   = tid >> 5;
    const int row0 = bt * 16;
    const int n0a  = wv * 32;
    const int n0b  = wv * 32 + 16;

    v8f accA = {};
    v8f accB = {};
    const __bf16* pBA = wT + (size_t)(n0a + (lane & 15)) * NF + (lane >> 4) * 16;
    const __bf16* pBB = wT + (size_t)(n0b + (lane & 15)) * NF + (lane >> 4) * 16;

    bf16x16 bA = *reinterpret_cast<const bf16x16*>(pBA);
    bf16x16 bB = *reinterpret_cast<const bf16x16*>(pBB);

    for (int k0 = 0; k0 < NF; k0 += 32) {
        __syncthreads();
        load_a_tile(ldsA, x4, row0, k0, tid);
        __syncthreads();
        bf16x16 a = *reinterpret_cast<const bf16x16*>(&ldsA[lane][0]);
        bf16x16 nbA = bA, nbB = bB;
        if (k0 + 32 < NF) {
            nbA = *reinterpret_cast<const bf16x16*>(pBA + k0 + 32);
            nbB = *reinterpret_cast<const bf16x16*>(pBB + k0 + 32);
        }
        accA = __builtin_amdgcn_wmma_f32_16x16x32_bf16(false, a, false, bA, (short)0, accA, false, false);
        accB = __builtin_amdgcn_wmma_f32_16x16x32_bf16(false, a, false, bB, (short)0, accB, false, false);
        bA = nbA; bB = nbB;
    }

    const int half = lane >> 4;
    const int cA = n0a + (lane & 15);
    const int cB = n0b + (lane & 15);
#pragma unroll
    for (int r = 0; r < 8; ++r) {
        int m = r + 8 * half;
        sH[m][cA] = fmaxf(accA[r], 0.f);
        sH[m][cB] = fmaxf(accB[r], 0.f);
    }
    __syncthreads();

    // logits: 160 dot products of length 256 (tiny K=10 tail, VALU is fine)
    if (tid < 16 * NC) {
        int m = tid / NC, c = tid % NC;
        float acc = 0.f;
        for (int k = 0; k < HID; ++k) acc += sH[m][k] * w1[k * NC + c];
        sL[m][c] = acc;
    }
    __syncthreads();

    if (tid < 16) {
        int m = tid;
        float l[NC];
        float mx = -1e30f;
        int   am = 0;
        for (int c = 0; c < NC; ++c) {
            l[c] = sL[m][c];
            if (l[c] > mx) { mx = l[c]; am = c; }   // first-occurrence argmax
        }
        float se = 0.f;
        for (int c = 0; c < NC; ++c) se += expf(l[c] - mx);
        float lse = mx + logf(se);
        int   yi  = y[row0 + m];
        atomicAdd(ce_sum, (double)(lse - l[yi]));
        atomicAdd(acc_sum, (am == yi) ? 1.0 : 0.0);
    }
}

// ---------------------------------------------------------------------------
// 5) per-class segment sums + counts + sum(x4^2), LDS-privatized
//    grid: 256 blocks x 256 threads; block handles BATCH/256 = 256 rows.
// ---------------------------------------------------------------------------
__global__ __launch_bounds__(256)
void segsum_kernel(const float* __restrict__ x4, const int* __restrict__ y,
                   float* __restrict__ gsums, float* __restrict__ gcounts,
                   double* __restrict__ gsumsq) {
    __shared__ float ls[NC * NF];     // 20 KB
    __shared__ float lc[NC];
    __shared__ float red[256];

    const int tid = threadIdx.x;
    for (int i = tid; i < NC * NF; i += 256) ls[i] = 0.f;
    if (tid < NC) lc[tid] = 0.f;
    __syncthreads();

    const int lane = tid & 31, wv = tid >> 5;
    const int rows_per_block = BATCH_SZ / gridDim.x;
    const int row0 = blockIdx.x * rows_per_block;

    float ssq = 0.f;
    for (int r = wv; r < rows_per_block; r += 8) {
        int i = row0 + r;
        int c = y[i];
        const float* xr = x4 + (size_t)i * NF;
        for (int k = lane; k < NF; k += 32) {
            float v = xr[k];
            ssq += v * v;
            atomicAdd(&ls[c * NF + k], v);     // ds_add_f32
        }
        if (lane == 0) atomicAdd(&lc[c], 1.0f);
    }

    red[tid] = ssq;
    __syncthreads();
    for (int o = 128; o > 0; o >>= 1) {
        if (tid < o) red[tid] += red[tid + o];
        __syncthreads();
    }

    for (int i = tid; i < NC * NF; i += 256) atomicAdd(&gsums[i], ls[i]);
    if (tid < NC) atomicAdd(&gcounts[tid], lc[tid]);
    if (tid == 0) atomicAdd(gsumsq, (double)red[0]);
}

// ---------------------------------------------------------------------------
// 6) finalize:  var = sumsq - sum_c ||s_c||^2 / max(n_c,1)
//               loss = ce/B + L1*l1 + ALPHA*var ;  acc = hits/B
// ---------------------------------------------------------------------------
__global__ __launch_bounds__(256)
void finalize_kernel(const float* __restrict__ sums, const float* __restrict__ counts,
                     const double* __restrict__ scal, float* __restrict__ out) {
    __shared__ double red[256];
    double s = 0.0;
    for (int i = threadIdx.x; i < NC * NF; i += 256) {
        int c = i / NF;
        double cnt = (double)fmaxf(counts[c], 1.0f);
        double sv  = (double)sums[i];
        s += sv * sv / cnt;
    }
    red[threadIdx.x] = s;
    __syncthreads();
    for (int o = 128; o > 0; o >>= 1) {
        if (threadIdx.x < o) red[threadIdx.x] += red[threadIdx.x + o];
        __syncthreads();
    }
    if (threadIdx.x == 0) {
        // scal: [0]=ce_sum [1]=acc_hits [2]=l1_sum [3]=sumsq
        double var  = scal[3] - red[0];
        double loss = scal[0] / (double)BATCH_SZ + L1_COEF * scal[2] + ALPHA * var;
        out[0] = (float)loss;
        out[1] = (float)(scal[1] / (double)BATCH_SZ);
    }
}

// ---------------------------------------------------------------------------
// launch
// ---------------------------------------------------------------------------
extern "C" void kernel_launch(void* const* d_in, const int* in_sizes, int n_in,
                              void* d_out, int out_size, void* d_ws, size_t ws_size,
                              hipStream_t stream) {
    const float* x   = (const float*)d_in[0];
    const int*   y   = (const int*)  d_in[1];
    const float* fc0 = (const float*)d_in[2];
    const float* fc1 = (const float*)d_in[3];
    const float* w   = (const float*)d_in[4];
    const float* w1  = (const float*)d_in[5];
    float* out = (float*)d_out;

    char* ws = (char*)d_ws;
    size_t off = 0;
    float*  x4    = (float*)(ws + off);  off += (size_t)BATCH_SZ * NF * sizeof(float); // 128 MB
    __bf16* fc0T  = (__bf16*)(ws + off); off += (size_t)NF * NF * sizeof(__bf16);
    __bf16* fc1T  = (__bf16*)(ws + off); off += (size_t)NF * NF * sizeof(__bf16);
    __bf16* wT    = (__bf16*)(ws + off); off += (size_t)NF * HID * sizeof(__bf16);
    float*  gsums = (float*)(ws + off);  off += (size_t)NC * NF * sizeof(float);
    float*  gcnts = (float*)(ws + off);  off += 64;
    double* scal  = (double*)(ws + off); off += 64;

    zero_kernel<<<20, 256, 0, stream>>>(gsums, gcnts, scal);

    // transpose+convert weights to bf16 [N][K], fused with L1 accumulation
    convT_l1_kernel<<<512, 256, 0, stream>>>(fc0, fc0T, NF, NF,  scal + 2);
    convT_l1_kernel<<<512, 256, 0, stream>>>(fc1, fc1T, NF, NF,  scal + 2);
    convT_l1_kernel<<<256, 256, 0, stream>>>(w,   wT,   NF, HID, scal + 2);
    l1_kernel<<<4, 256, 0, stream>>>(w1, HID * NC, scal + 2);

    dim3 g1(BATCH_SZ / 16, 4);
    gemm_gate_kernel<<<g1, 256, 0, stream>>>(x, fc0T, fc1T, x4);

    head_kernel<<<BATCH_SZ / 16, 256, 0, stream>>>(x4, wT, w1, y, scal + 0, scal + 1);

    segsum_kernel<<<256, 256, 0, stream>>>(x4, y, gsums, gcnts, scal + 3);

    finalize_kernel<<<1, 256, 0, stream>>>(gsums, gcnts, scal, out);
}
